// HistogramLoss_77558519431866
// MI455X (gfx1250) — matrix operand order repaired and checked by reading, not compiled, not run on hardware
//
#include <hip/hip_runtime.h>
#include <hip/hip_bf16.h>

// ---------------------------------------------------------------------------
// HistogramLoss for MI455X (gfx1250, wave32).
//
// Math reduction used:
//   * soft-hist activation is nonzero for exactly one bin per pixel
//     (|x-c| < delta/2 window == the bin the pixel falls in).
//   * sum-over-bins of cumsum differences == dot(64-j, h1-h2).
// => one streaming histogram pass + one tiny WMMA-based weighted reduction.
// ---------------------------------------------------------------------------

#define NUM_BINS 64
#define HW (256 * 256)
#define NBATCH 8

__device__ __constant__ const float EDGE0     = -0.05f;
__device__ __constant__ const float DELTA     = 1.1f / 64.0f;        // 0.0171875
__device__ __constant__ const float INV_DELTA = 64.0f / 1.1f;
__device__ __constant__ const float HWIDTH    = 0.5f * (1.1f / 64.0f);
__device__ __constant__ const float LOG2_1P01 = 0.014355292977070041f; // log2(1.01)
__device__ __constant__ const float INV_HW    = 1.0f / (float)HW;

// scratch layout in d_ws: float hist[4][8][3][64]
//   set 0 = rgb(input), 1 = rgb(ref), 2 = yuv(input), 3 = yuv(ref)
//   flat idx = set*1536 + b*192 + c*64 + j
#define HIST_FLOATS (4 * NBATCH * 3 * NUM_BINS)

__global__ __launch_bounds__(256) void zero_ws(float* __restrict__ gh, int n) {
    int i = blockIdx.x * blockDim.x + threadIdx.x;
    if (i < n) gh[i] = 0.0f;
}

__device__ __forceinline__ void bin_accum(float x, float* __restrict__ lhc) {
    float xs = (x + 1.0f) * 0.5f;                       // [-1,1] -> [0,1]
    float t  = (xs - EDGE0) * INV_DELTA;
    int   j  = (int)floorf(t);
    if (j >= 0 && j < NUM_BINS) {
        float c   = EDGE0 + ((float)j + 0.5f) * DELTA;  // bin center
        float d   = fabsf(xs - c);
        float act = exp2f((HWIDTH - d) * LOG2_1P01);    // 1.01^(width-d)
        if (act > 1.0f) atomicAdd(lhc + j, act);        // ds_add_f32
    }
}

__device__ __forceinline__ void accum_px(float r, float g, float b,
                                         int set_rgb, int set_yuv,
                                         float* __restrict__ lh) {
    bin_accum(r, lh + (set_rgb * 3 + 0) * NUM_BINS);
    bin_accum(g, lh + (set_rgb * 3 + 1) * NUM_BINS);
    bin_accum(b, lh + (set_rgb * 3 + 2) * NUM_BINS);
    float y =  0.299f   * r + 0.587f   * g + 0.114f   * b;
    float u = -0.14713f * r - 0.28886f * g + 0.436f   * b;
    float v =  0.615f   * r - 0.51499f * g - 0.10001f * b;
    bin_accum(y, lh + (set_yuv * 3 + 0) * NUM_BINS);
    bin_accum(u, lh + (set_yuv * 3 + 1) * NUM_BINS);
    bin_accum(v, lh + (set_yuv * 3 + 2) * NUM_BINS);
}

// Streaming pass: b128 loads of 4 pixels x 3 channels x 2 images, per-wave
// LDS histograms (8 waves x 768 floats = 24 KB of the 320 KB WGP LDS).
__global__ __launch_bounds__(256) void hist_kernel(const float* __restrict__ img_in,
                                                   const float* __restrict__ img_ref,
                                                   float* __restrict__ gh) {
    __shared__ float lh[8 * 4 * 3 * NUM_BINS];           // [wave][set][c][bin]
    for (int i = threadIdx.x; i < 8 * 768; i += blockDim.x) lh[i] = 0.0f;
    __syncthreads();

    const int wid = threadIdx.x >> 5;                    // wave32
    float* mylh = lh + wid * 768;

    const int b = blockIdx.y;
    const float4* in4  = (const float4*)(img_in  + (size_t)b * 3 * HW);
    const float4* ref4 = (const float4*)(img_ref + (size_t)b * 3 * HW);
    const int NQ = HW / 4;

    for (int q = blockIdx.x * blockDim.x + threadIdx.x; q < NQ;
         q += gridDim.x * blockDim.x) {
        float4 ir = in4[0 * NQ + q];
        float4 ig = in4[1 * NQ + q];
        float4 ib = in4[2 * NQ + q];
        float4 rr = ref4[0 * NQ + q];
        float4 rg = ref4[1 * NQ + q];
        float4 rb = ref4[2 * NQ + q];
        float irx[4] = {ir.x, ir.y, ir.z, ir.w};
        float igx[4] = {ig.x, ig.y, ig.z, ig.w};
        float ibx[4] = {ib.x, ib.y, ib.z, ib.w};
        float rrx[4] = {rr.x, rr.y, rr.z, rr.w};
        float rgx[4] = {rg.x, rg.y, rg.z, rg.w};
        float rbx[4] = {rb.x, rb.y, rb.z, rb.w};
#pragma unroll
        for (int e = 0; e < 4; ++e) {
            accum_px(irx[e], igx[e], ibx[e], 0, 2, mylh);
            accum_px(rrx[e], rgx[e], rbx[e], 1, 3, mylh);
        }
    }
    __syncthreads();

    // merge 8 wave-private hists, then a few global fp32 atomics into L2 scratch
    for (int i = threadIdx.x; i < 768; i += blockDim.x) {
        float s = 0.0f;
#pragma unroll
        for (int w = 0; w < 8; ++w) s += lh[w * 768 + i];
        int set = i / 192, rem = i % 192;                // rem = c*64 + j
        atomicAdd(&gh[set * 1536 + b * 192 + rem], s);
    }
}

// ---------------------------------------------------------------------------
// Finalize: loss = (1/8) * sum over 48 channels of dot(w_j = 64-j, delta_j)
// via V_WMMA_F32_16X16X32_F16. A rows = channel deltas, B col 0 = weights.
// One wave, 3 channel groups x 2 K-chunks accumulate into one v8f C.
// ---------------------------------------------------------------------------
typedef __attribute__((ext_vector_type(16))) _Float16 v16h;
typedef __attribute__((ext_vector_type(8)))  float    v8f;

__global__ __launch_bounds__(32) void emd_kernel(const float* __restrict__ gh,
                                                 float* __restrict__ out) {
    const int lane = threadIdx.x;        // 0..31
    const int m    = lane & 15;          // A row / B column / D column
    const int hi   = lane >> 4;          // lane half select

    v8f acc = {};
#pragma unroll
    for (int g = 0; g < 3; ++g) {
        int ch       = g * 16 + m;                 // 0..47
        int pairbase = (ch < 24) ? 0 : 2;          // rgb sets (0,1) / yuv (2,3)
        int local    = (ch < 24) ? ch : ch - 24;   // b*3 + c
        const float* hin  = gh + pairbase * 1536 + local * NUM_BINS;
        const float* href = gh + (pairbase + 1) * 1536 + local * NUM_BINS;
#pragma unroll
        for (int kb = 0; kb < NUM_BINS; kb += 32) {
            v16h a, bm;
            // A 16x32 f16 layout (ISA 7.12.2): lane half hi holds
            //   elems 0..7  -> K = hi*8 + e,  elems 8..15 -> K = 16 + hi*8 + e
#pragma unroll
            for (int e = 0; e < 8; ++e) {
                int k0 = kb + hi * 8 + e;
                int k1 = kb + 16 + hi * 8 + e;
                a[e]     = (_Float16)((hin[k0] - href[k0]) * INV_HW);
                a[8 + e] = (_Float16)((hin[k1] - href[k1]) * INV_HW);
            }
            // B 32x16 f16: lane column n = m, same K order as A (e<8: hi*8+e, else 16+hi*8+e-8).
            // Only column 0 carries the EMD weights (64 - k); rest zero.
#pragma unroll
            for (int e = 0; e < 16; ++e) {
                int k = kb + ((e < 8) ? (hi * 8 + e) : (16 + hi * 8 + (e - 8)));
                bm[e] = (_Float16)((m == 0) ? (float)(NUM_BINS - k) : 0.0f);
            }
            acc = __builtin_amdgcn_wmma_f32_16x16x32_f16(
                false, a, false, bm, (short)0, acc, false, false);
        }
    }

    // D[M][0] lives in lanes 0 (M=0..7) and 16 (M=8..15); sum and wave-reduce.
    float s = 0.0f;
    if (m == 0) {
#pragma unroll
        for (int v = 0; v < 8; ++v) s += acc[v];
    }
#pragma unroll
    for (int off = 16; off > 0; off >>= 1) s += __shfl_down(s, off, 32);
    if (lane == 0) out[0] = s * (1.0f / (float)NBATCH);
}

extern "C" void kernel_launch(void* const* d_in, const int* in_sizes, int n_in,
                              void* d_out, int out_size, void* d_ws, size_t ws_size,
                              hipStream_t stream) {
    const float* img_in  = (const float*)d_in[0];
    const float* img_ref = (const float*)d_in[1];
    float*       out     = (float*)d_out;
    float*       gh      = (float*)d_ws;   // needs HIST_FLOATS*4 = 24 KB

    zero_ws<<<dim3((HIST_FLOATS + 255) / 256), dim3(256), 0, stream>>>(gh, HIST_FLOATS);
    hist_kernel<<<dim3(32, NBATCH), dim3(256), 0, stream>>>(img_in, img_ref, gh);
    emd_kernel<<<dim3(1), dim3(32), 0, stream>>>(gh, out);
    (void)in_sizes; (void)n_in; (void)out_size; (void)ws_size;
}
